// NemotronHMambaDecoderLayer_83236466196647
// MI455X (gfx1250) — compile-verified, matrix operand order checked by reading, not run on hardware
//
#include <hip/hip_runtime.h>

typedef unsigned short u16;
typedef unsigned int   u32;
typedef __attribute__((ext_vector_type(8)))  __bf16 v8bf;
typedef __attribute__((ext_vector_type(16))) __bf16 v16bf;
typedef __attribute__((ext_vector_type(8)))  float  v8f;

namespace cfg {
constexpr int Bsz = 2, L = 1024, Dm = 2048;
constexpr int G = 8, Nst = 128, H = 64, P = 64, Kc = 4, R = H / G;     // R=8
constexpr int d_inner  = 2 * Dm;                                       // 4096
constexpr int conv_dim = d_inner + 2 * G * Nst;                        // 6144
constexpr int proj_dim = 2 * d_inner + 2 * G * Nst + H;                // 10304
constexpr int Mrows    = Bsz * L;                                      // 2048
}

__device__ __forceinline__ u16 f32_to_bf16(float f) {
  u32 u = __float_as_uint(f);
  u = (u + 0x7FFFu + ((u >> 16) & 1u)) >> 16;   // round-to-nearest-even
  return (u16)u;
}
__device__ __forceinline__ float sigmoidf_(float x) { return 1.0f / (1.0f + __expf(-x)); }

// CDNA5 async memory->LDS DMA (ASYNCcnt-tracked); lds_off is the wave-relative
// LDS byte address (low 32 bits of a generic pointer to a __shared__ object).
__device__ __forceinline__ void async_copy_b128(u32 lds_off, const void* gaddr) {
  asm volatile("global_load_async_to_lds_b128 %0, %1, off"
               :: "v"(lds_off), "v"(gaddr) : "memory");
}
__device__ __forceinline__ void wait_async_le3() {
  asm volatile("s_wait_asynccnt 0x3" ::: "memory");
}
__device__ __forceinline__ void wait_async_0() {
  asm volatile("s_wait_asynccnt 0x0" ::: "memory");
}

// ---------------- fp32 -> bf16 bulk convert (weights) ----------------
__global__ __launch_bounds__(256)
void k_cvt_bf16(const float* __restrict__ in, u16* __restrict__ out, long n) {
  long i = (long)blockIdx.x * blockDim.x + threadIdx.x;
  long stride = (long)gridDim.x * blockDim.x;
  for (; i < n; i += stride) out[i] = f32_to_bf16(in[i]);
}

// ---------------- residual add + RMSNorm -> new_residual (f32) + x (bf16) ----------------
__global__ __launch_bounds__(256)
void k_add_rmsnorm(const float* __restrict__ hs, const float* __restrict__ res,
                   const float* __restrict__ w, float* __restrict__ resid_out,
                   u16* __restrict__ xbf) {
  using namespace cfg;
  int row = blockIdx.x, tid = threadIdx.x;
  const float* ph = hs + (size_t)row * Dm;
  const float* pr = res + (size_t)row * Dm;
  constexpr int PT = Dm / 256;  // 8
  float loc[PT];
  float ss = 0.f;
#pragma unroll
  for (int i = 0; i < PT; ++i) {
    int e = tid + i * 256;
    float v = ph[e] + pr[e];
    loc[i] = v;
    resid_out[(size_t)row * Dm + e] = v;
    ss += v * v;
  }
  __shared__ float red[256];
  red[tid] = ss; __syncthreads();
  for (int s = 128; s > 0; s >>= 1) { if (tid < s) red[tid] += red[tid + s]; __syncthreads(); }
  float scale = rsqrtf(red[0] / (float)Dm + 1e-5f);
#pragma unroll
  for (int i = 0; i < PT; ++i) {
    int e = tid + i * 256;
    xbf[(size_t)row * Dm + e] = f32_to_bf16(loc[i] * scale * w[e]);
  }
}

// ---------------- bf16 NT GEMM: C[M,N] = A[M,K] * W[N,K]^T, f32 accumulate ----------------
// block tile 128x64, 8 waves as 4(M) x 2(N), each wave 32x32 (4 wmma accums), BK=32.
// Double-buffered LDS fed by global_load_async_to_lds_b128 (ASYNCcnt pipeline).
__global__ __launch_bounds__(256)
void k_gemm_bf16nt(const u16* __restrict__ A, const u16* __restrict__ W,
                   float* __restrict__ C, int M, int Nn, int Kk) {
  __shared__ __align__(16) u16 As[2][128][40];   // +8 pad: 80B row stride, 16B aligned
  __shared__ __align__(16) u16 Bs[2][64][40];
  int tid = threadIdx.x;
  int m0 = blockIdx.y * 128, n0 = blockIdx.x * 64;
  int lane = tid & 31, lr = lane & 15, kh = lane >> 4;
  int wv = tid >> 5;
  int wm = (wv & 3) * 32, wn = (wv >> 2) * 32;
  v8f acc[2][2] = {};

  // per-thread staging slots (16B chunks): A rows r and r+64, W row r (cols cc..cc+7)
  int r  = tid >> 2;
  int cc = (tid & 3) << 3;

  auto issue = [&](int st, int k0) {
    async_copy_b128((u32)(size_t)&As[st][r][cc],      A + (size_t)(m0 + r) * Kk + k0 + cc);
    async_copy_b128((u32)(size_t)&As[st][r + 64][cc], A + (size_t)(m0 + r + 64) * Kk + k0 + cc);
    async_copy_b128((u32)(size_t)&Bs[st][r][cc],      W + (size_t)(n0 + r) * Kk + k0 + cc);
  };

  int nk = Kk >> 5;
  issue(0, 0);
  for (int t = 0; t < nk; ++t) {
    int st = t & 1;
    if (t + 1 < nk) { issue(st ^ 1, (t + 1) << 5); wait_async_le3(); }
    else            { wait_async_0(); }
    __syncthreads();   // all waves' tile-t DMA landed in LDS

    // fragment loads per ISA 16-bit A/B layout: lane(lr,kh) holds two
    // contiguous 8-elem K runs at K = 8*kh and K = 16+8*kh
    v16bf af[2], bfv[2];
#pragma unroll
    for (int mi = 0; mi < 2; ++mi) {
      const v8bf* p = (const v8bf*)&As[st][wm + mi * 16 + lr][0];
      v8bf lo = p[kh], hi = p[2 + kh];
      af[mi] = __builtin_shufflevector(lo, hi, 0,1,2,3,4,5,6,7,8,9,10,11,12,13,14,15);
    }
#pragma unroll
    for (int ni = 0; ni < 2; ++ni) {
      const v8bf* p = (const v8bf*)&Bs[st][wn + ni * 16 + lr][0];
      v8bf lo = p[kh], hi = p[2 + kh];
      bfv[ni] = __builtin_shufflevector(lo, hi, 0,1,2,3,4,5,6,7,8,9,10,11,12,13,14,15);
    }
#pragma unroll
    for (int mi = 0; mi < 2; ++mi)
#pragma unroll
      for (int ni = 0; ni < 2; ++ni)
        acc[mi][ni] = __builtin_amdgcn_wmma_f32_16x16x32_bf16(
            false, af[mi], false, bfv[ni], (short)0, acc[mi][ni], false, false);
    __syncthreads();   // buffer st free for tile t+2's DMA next iteration
  }

  // epilogue: C/D layout -> lane lr = N, VGPR v -> M = v + 8*kh
#pragma unroll
  for (int mi = 0; mi < 2; ++mi)
#pragma unroll
    for (int ni = 0; ni < 2; ++ni) {
      int n = n0 + wn + ni * 16 + lr;
#pragma unroll
      for (int v = 0; v < 8; ++v) {
        int m = m0 + wm + mi * 16 + v + 8 * kh;
        C[(size_t)m * Nn + n] = acc[mi][ni][v];
      }
    }
}

// ---------------- depthwise causal conv1d (K=4) + SiLU ----------------
__global__ __launch_bounds__(256)
void k_conv_silu(const float* __restrict__ proj, const float* __restrict__ cw,
                 const float* __restrict__ cb, float* __restrict__ xconv) {
  using namespace cfg;
  long idx = (long)blockIdx.x * 256 + threadIdx.x;
  long total = (long)Mrows * conv_dim;
  if (idx >= total) return;
  int c  = (int)(idx % conv_dim);
  int rl = (int)(idx / conv_dim);
  int l  = rl % L;
  float acc = cb[c];
#pragma unroll
  for (int k = 0; k < Kc; ++k) {
    int lk = l + k - (Kc - 1);
    if (lk >= 0)
      acc += proj[(size_t)(rl + k - (Kc - 1)) * proj_dim + d_inner + c] * cw[c * Kc + k];
  }
  acc = acc * sigmoidf_(acc);
  xconv[(size_t)rl * conv_dim + c] = acc;
}

// ---------------- dt = softplus(dt + bias); dA = exp(dt * -exp(A_log)) ----------------
__global__ __launch_bounds__(256)
void k_dt_prep(const float* __restrict__ proj, const float* __restrict__ A_log,
               const float* __restrict__ dt_bias, float* __restrict__ dtv,
               float* __restrict__ dA) {
  using namespace cfg;
  int idx = blockIdx.x * 256 + threadIdx.x;
  if (idx >= Mrows * H) return;
  int h = idx % H, row = idx / H;
  float x = proj[(size_t)row * proj_dim + d_inner + conv_dim + h] + dt_bias[h];
  float sp = (x > 20.f) ? x : log1pf(__expf(x));
  float Av = -__expf(A_log[h]);
  dtv[idx] = sp;
  dA[idx]  = __expf(sp * Av);
}

// ---------------- selective scan: block = (b,g,r), lane = p, state N=128 in VGPRs ----------------
__global__ __launch_bounds__(64)
void k_scan(const float* __restrict__ xconv, const float* __restrict__ dtv,
            const float* __restrict__ dAv, const float* __restrict__ Dp_,
            float* __restrict__ ys) {
  using namespace cfg;
  int blk = blockIdx.x;
  int b = blk / (G * R);
  int g = (blk / R) % G;
  int r = blk % R;
  int hh = g * R + r;
  int p = threadIdx.x;
  float Dp = Dp_[hh];
  float hstate[Nst];
#pragma unroll
  for (int n = 0; n < Nst; ++n) hstate[n] = 0.f;
  __shared__ float Bsh[Nst], Csh[Nst];
  for (int t = 0; t < L; ++t) {
    int row = b * L + t;
    const float* xr = xconv + (size_t)row * conv_dim;
    float da = dAv[(size_t)row * H + hh];
    float dt = dtv[(size_t)row * H + hh];
    float xt = xr[hh * P + p];
    Bsh[p]      = xr[d_inner + g * Nst + p];
    Bsh[p + 64] = xr[d_inner + g * Nst + p + 64];
    Csh[p]      = xr[d_inner + G * Nst + g * Nst + p];
    Csh[p + 64] = xr[d_inner + G * Nst + g * Nst + p + 64];
    __syncthreads();
    float dbx = dt * xt;
    float y = 0.f;
#pragma unroll
    for (int n = 0; n < Nst; ++n) {
      hstate[n] = hstate[n] * da + dbx * Bsh[n];
      y += hstate[n] * Csh[n];
    }
    __syncthreads();
    ys[(size_t)row * d_inner + hh * P + p] = y + Dp * xt;
  }
}

// ---------------- gated RMSNorm: rmsnorm(ys * silu(z)) * gw -> bf16 ----------------
__global__ __launch_bounds__(256)
void k_gate_norm(const float* __restrict__ proj, const float* __restrict__ ys,
                 const float* __restrict__ gw, u16* __restrict__ ybf) {
  using namespace cfg;
  int row = blockIdx.x, tid = threadIdx.x;
  constexpr int PT = d_inner / 256;  // 16
  float loc[PT];
  float ss = 0.f;
#pragma unroll
  for (int i = 0; i < PT; ++i) {
    int e = tid + i * 256;
    float z = proj[(size_t)row * proj_dim + e];
    float v = ys[(size_t)row * d_inner + e] * (z * sigmoidf_(z));
    loc[i] = v; ss += v * v;
  }
  __shared__ float red[256];
  red[tid] = ss; __syncthreads();
  for (int s = 128; s > 0; s >>= 1) { if (tid < s) red[tid] += red[tid + s]; __syncthreads(); }
  float scale = rsqrtf(red[0] / (float)d_inner + 1e-5f);
#pragma unroll
  for (int i = 0; i < PT; ++i) {
    int e = tid + i * 256;
    ybf[(size_t)row * d_inner + e] = f32_to_bf16(loc[i] * scale * gw[e]);
  }
}

extern "C" void kernel_launch(void* const* d_in, const int* in_sizes, int n_in,
                              void* d_out, int out_size, void* d_ws, size_t ws_size,
                              hipStream_t stream) {
  using namespace cfg;
  (void)in_sizes; (void)n_in; (void)out_size; (void)ws_size;

  const float* hidden     = (const float*)d_in[0];
  const float* residual   = (const float*)d_in[1];
  const float* norm_w     = (const float*)d_in[2];
  const float* in_proj_w  = (const float*)d_in[3];
  const float* conv_w     = (const float*)d_in[4];
  const float* conv_b     = (const float*)d_in[5];
  const float* A_log      = (const float*)d_in[6];
  const float* D_param    = (const float*)d_in[7];
  const float* dt_bias    = (const float*)d_in[8];
  const float* gate_norm_w= (const float*)d_in[9];
  const float* out_proj_w = (const float*)d_in[10];

  float* out       = (float*)d_out;                        // (B,L,Dm)
  float* resid_out = out + (size_t)Mrows * Dm;             // (B,L,Dm)

  char* ws = (char*)d_ws;
  size_t off = 0;
  auto alloc = [&](size_t bytes) { size_t o = off; off = (off + bytes + 255) & ~(size_t)255; return o; };
  u16*   xbf   = (u16*)  (ws + alloc((size_t)Mrows * Dm * 2));
  u16*   winb  = (u16*)  (ws + alloc((size_t)proj_dim * Dm * 2));
  u16*   woutb = (u16*)  (ws + alloc((size_t)Dm * d_inner * 2));
  float* proj  = (float*)(ws + alloc((size_t)Mrows * proj_dim * 4));
  float* xconv = (float*)(ws + alloc((size_t)Mrows * conv_dim * 4));
  float* dtv   = (float*)(ws + alloc((size_t)Mrows * H * 4));
  float* dAv   = (float*)(ws + alloc((size_t)Mrows * H * 4));
  float* ysv   = (float*)(ws + alloc((size_t)Mrows * d_inner * 4));
  u16*   ybf   = (u16*)  (ws + alloc((size_t)Mrows * d_inner * 2));

  // weights fp32 -> bf16 (once per call; cheap, memory-bound)
  k_cvt_bf16<<<2048, 256, 0, stream>>>(in_proj_w,  winb,  (long)proj_dim * Dm);
  k_cvt_bf16<<<2048, 256, 0, stream>>>(out_proj_w, woutb, (long)Dm * d_inner);

  // residual add + RMSNorm
  k_add_rmsnorm<<<Mrows, 256, 0, stream>>>(hidden, residual, norm_w, resid_out, xbf);

  // in_proj: (2048 x 2048) @ (10304 x 2048)^T -> proj (2048 x 10304)
  k_gemm_bf16nt<<<dim3(proj_dim / 64, Mrows / 128), 256, 0, stream>>>(
      xbf, winb, proj, Mrows, proj_dim, Dm);

  // causal depthwise conv + SiLU on the xBC slice
  k_conv_silu<<<((long)Mrows * conv_dim + 255) / 256, 256, 0, stream>>>(
      proj, conv_w, conv_b, xconv);

  // dt softplus + dA precompute
  k_dt_prep<<<(Mrows * H + 255) / 256, 256, 0, stream>>>(proj, A_log, dt_bias, dtv, dAv);

  // selective scan
  k_scan<<<Bsz * G * R, 64, 0, stream>>>(xconv, dtv, dAv, D_param, ysv);

  // gated RMSNorm -> bf16
  k_gate_norm<<<Mrows, 256, 0, stream>>>(proj, ysv, gate_norm_w, ybf);

  // out_proj: (2048 x 4096) @ (2048 x 4096)^T -> out (2048 x 2048)
  k_gemm_bf16nt<<<dim3(Dm / 64, Mrows / 128), 256, 0, stream>>>(
      ybf, woutb, out, Mrows, Dm, d_inner);
}
